// CBM_4002909520795
// MI455X (gfx1250) — compile-verified, matrix operand order checked by reading, not run on hardware
//
#include <hip/hip_runtime.h>

// ---------------------------------------------------------------------------
// CDNA5 (gfx1250) implementation of the concept-bottleneck decoder reference.
// All GEMMs use V_WMMA_F32_16X16X32_BF16 (bf16 operands, f32 accumulate).
// Reductions (layernorm, softmax, L2 norms) stay in f32.
// Self-attention on a length-1 target is exactly Wo(Wv t + bv) + bo, so the
// sa.Wq/Wk/bq/bk parameters are provably unused.
// ---------------------------------------------------------------------------

typedef __bf16 bf16_t;
typedef __attribute__((ext_vector_type(16))) __bf16 v16bf;
typedef __attribute__((ext_vector_type(8)))  __bf16 v8bf;
typedef __attribute__((ext_vector_type(8)))  float  v8f;

union FragU { v16bf v; v8bf h[2]; };

#define BB    4096
#define DMODEL 512
#define NCONC 10000
#define TOPK  50
#define LKV   51      // 1 + TOPK
#define DFF   2048
#define NLAYER 6

__device__ __forceinline__ float b2f(bf16_t b) {
  unsigned short s = __builtin_bit_cast(unsigned short, b);
  unsigned u = ((unsigned)s) << 16;
  return __builtin_bit_cast(float, u);
}
__device__ __forceinline__ bf16_t f2b(float f) {
  unsigned u = __builtin_bit_cast(unsigned, f);
  unsigned r = u + 0x7FFFu + ((u >> 16) & 1u);   // round-to-nearest-even
  unsigned short s = (unsigned short)(r >> 16);
  return __builtin_bit_cast(bf16_t, s);
}

// ---------------------------------------------------------------------------
// Generic GEMM:  C[M,N] = A[M,K] * W[N,K]^T (+ bias) (+relu)
// A, W bf16 row-major (K contiguous for BOTH operands -> two b128 loads per
// lane per fragment, matching the 16-bit WMMA VGPR layout exactly).
// Wave computes a 16(M) x 64(N) strip (A fragment reused 4x). Block = 8 waves
// = 128 rows. Requires M % 128 == 0 (true for all call sites: 4096, 4096*51).
// N may be ragged (score GEMM N=10000): loads clamp, stores guard.
// ---------------------------------------------------------------------------
__global__ __launch_bounds__(256) void wmma_gemm_bf16(
    const bf16_t* __restrict__ A, const bf16_t* __restrict__ W,
    const float* __restrict__ bias, float* __restrict__ Cf,
    bf16_t* __restrict__ Cb, int M, int N, int K, int relu)
{
  const int wave = threadIdx.x >> 5;
  const int lane = threadIdx.x & 31;
  const int r    = lane & 15;     // A row within tile / B column within tile
  const int hh   = lane >> 4;     // half-select (K groups, and C row group)
  const int m0 = (blockIdx.y * 8 + wave) << 4;
  const int n0 = blockIdx.x << 6;
  if (m0 >= M) return;

  v8f acc[4] = {};

  const bf16_t* arow = A + (size_t)(m0 + r) * K;
  const bf16_t* wrow[4];
#pragma unroll
  for (int s = 0; s < 4; ++s) {
    int n = n0 + s * 16 + r;
    wrow[s] = W + (size_t)(n < N ? n : 0) * K;   // clamp; stores are guarded
  }

  for (int k0 = 0; k0 < K; k0 += 32) {
    FragU a;
    a.h[0] = *(const v8bf*)(arow + k0 + hh * 8);
    a.h[1] = *(const v8bf*)(arow + k0 + 16 + hh * 8);
    if (k0 + 64 < K) __builtin_prefetch(arow + k0 + 64, 0, 1);
#pragma unroll
    for (int s = 0; s < 4; ++s) {
      FragU b;
      b.h[0] = *(const v8bf*)(wrow[s] + k0 + hh * 8);
      b.h[1] = *(const v8bf*)(wrow[s] + k0 + 16 + hh * 8);
      acc[s] = __builtin_amdgcn_wmma_f32_16x16x32_bf16(
          false, a.v, false, b.v, (short)0, acc[s], false, false);
    }
  }

#pragma unroll
  for (int s = 0; s < 4; ++s) {
    int n = n0 + s * 16 + r;
    if (n >= N) continue;
    float bv = bias ? bias[n] : 0.0f;
#pragma unroll
    for (int v = 0; v < 8; ++v) {
      int m = m0 + v + 8 * hh;            // C/D layout: VGPR v, lane half hh
      float val = acc[s][v] + bv;
      if (relu) val = fmaxf(val, 0.0f);
      size_t o = (size_t)m * N + n;
      if (Cf) Cf[o] = val;
      if (Cb) Cb[o] = f2b(val);
    }
  }
}

// ---------------------------------------------------------------------------
// Per-row top-50 (sorted descending, jax tie-break = lowest index) + softmax
// of the selected scores. One block (256 threads) per row; 40 strided values
// per thread held in registers; 50 iterative argmax block-reductions.
// ---------------------------------------------------------------------------
__global__ __launch_bounds__(256) void topk_softmax(
    const float* __restrict__ scores, int* __restrict__ idx_out,
    float* __restrict__ w_out)
{
  __shared__ float rv[256];
  __shared__ int   ri[256];
  __shared__ float sel[TOPK];
  const int row = blockIdx.x, t = threadIdx.x;
  const float* sr = scores + (size_t)row * NCONC;

  float vals[40];
#pragma unroll
  for (int i = 0; i < 40; ++i) {
    int c = t + i * 256;
    vals[i] = (c < NCONC) ? sr[c] : -__builtin_inff();
  }

  for (int k = 0; k < TOPK; ++k) {
    float bv = -__builtin_inff(); int bi = 0x7fffffff;
#pragma unroll
    for (int i = 0; i < 40; ++i) {
      int c = t + i * 256;
      if (vals[i] > bv) { bv = vals[i]; bi = c; }
    }
    rv[t] = bv; ri[t] = bi;
    __syncthreads();
    for (int s = 128; s > 0; s >>= 1) {
      if (t < s) {
        if (rv[t + s] > rv[t] ||
            (rv[t + s] == rv[t] && ri[t + s] < ri[t])) {
          rv[t] = rv[t + s]; ri[t] = ri[t + s];
        }
      }
      __syncthreads();
    }
    const float wv = rv[0]; const int wi = ri[0];
#pragma unroll
    for (int i = 0; i < 40; ++i)
      if (t + i * 256 == wi) vals[i] = -__builtin_inff();
    if (t == 0) { sel[k] = wv; idx_out[(size_t)row * TOPK + k] = wi; }
    __syncthreads();
  }

  if (t == 0) {
    float m = sel[0];
    for (int k = 1; k < TOPK; ++k) m = fmaxf(m, sel[k]);
    float e[TOPK]; float sum = 0.0f;
    for (int k = 0; k < TOPK; ++k) { e[k] = expf(sel[k] - m); sum += e[k]; }
    float inv = 1.0f / sum;
    for (int k = 0; k < TOPK; ++k) w_out[(size_t)row * TOPK + k] = e[k] * inv;
  }
}

// kv[r,0] = x[r] + type_emb[0];  kv[r,1+k] = w*concept[idx] + pos[k] + type[1]
__global__ __launch_bounds__(256) void build_kv(
    const float* __restrict__ x, const float* __restrict__ cf,
    const float* __restrict__ type_emb, const float* __restrict__ pos_emb,
    const int* __restrict__ idx, const float* __restrict__ w,
    bf16_t* __restrict__ kvb)
{
  const int tok = blockIdx.x;
  const int r = tok / LKV, j = tok % LKV, t = threadIdx.x;
  const size_t base = (size_t)tok * DMODEL;
#pragma unroll
  for (int i = 0; i < 2; ++i) {
    int c = t + i * 256;
    float v;
    if (j == 0) {
      v = x[(size_t)r * DMODEL + c] + type_emb[c];
    } else {
      int k  = j - 1;
      int ci = idx[(size_t)r * TOPK + k];
      v = w[(size_t)r * TOPK + k] * cf[(size_t)ci * DMODEL + c] +
          pos_emb[(size_t)k * DMODEL + c] + type_emb[DMODEL + c];
    }
    kvb[base + c] = f2b(v);
  }
}

__global__ __launch_bounds__(256) void broadcast_cls(
    const float* __restrict__ cls, float* __restrict__ tf,
    bf16_t* __restrict__ tb)
{
  const size_t base = (size_t)blockIdx.x * DMODEL;
  const int t = threadIdx.x;
#pragma unroll
  for (int i = 0; i < 2; ++i) {
    int c = t + i * 256; float v = cls[c];
    tf[base + c] = v; tb[base + c] = f2b(v);
  }
}

__global__ __launch_bounds__(256) void f32_to_bf16(
    const float* __restrict__ in, bf16_t* __restrict__ out, long n)
{
  long i = (long)blockIdx.x * 256 + threadIdx.x;
  if (i < n) out[i] = f2b(in[i]);
}

// LayerNorm(res + x) * g + b ; writes f32 (may alias `res`) and bf16 copy.
__global__ __launch_bounds__(256) void ln_residual(
    const float* __restrict__ xin, const float* __restrict__ res,
    const float* __restrict__ g, const float* __restrict__ b,
    float* __restrict__ outf, bf16_t* __restrict__ outb)
{
  __shared__ float red[256], red2[256];
  const int row = blockIdx.x, t = threadIdx.x;
  const size_t base = (size_t)row * DMODEL;
  float v0 = xin[base + t] + res[base + t];
  float v1 = xin[base + t + 256] + res[base + t + 256];
  red[t] = v0 + v1; red2[t] = v0 * v0 + v1 * v1;
  __syncthreads();
  for (int s = 128; s > 0; s >>= 1) {
    if (t < s) { red[t] += red[t + s]; red2[t] += red2[t + s]; }
    __syncthreads();
  }
  const float mean = red[0] * (1.0f / DMODEL);
  const float var  = red2[0] * (1.0f / DMODEL) - mean * mean;
  const float rstd = rsqrtf(var + 1e-5f);
  float o0 = (v0 - mean) * rstd * g[t] + b[t];
  float o1 = (v1 - mean) * rstd * g[t + 256] + b[t + 256];
  outf[base + t] = o0; outf[base + t + 256] = o1;
  if (outb) { outb[base + t] = f2b(o0); outb[base + t + 256] = f2b(o1); }
}

// Cross attention, 1 query vs 51 keys, 8 heads of 64. Block per row.
__global__ __launch_bounds__(256) void cross_attn(
    const float* __restrict__ qf, const bf16_t* __restrict__ kh,
    const bf16_t* __restrict__ vh, bf16_t* __restrict__ ob)
{
  __shared__ float qs[DMODEL];
  __shared__ float lg[8 * LKV];
  const int row = blockIdx.x, t = threadIdx.x;
  qs[t]       = qf[(size_t)row * DMODEL + t];
  qs[t + 256] = qf[(size_t)row * DMODEL + t + 256];
  __syncthreads();

  if (t < 8 * LKV) {
    const int h = t / LKV, j = t % LKV;
    const bf16_t* kr = kh + ((size_t)row * LKV + j) * DMODEL + h * 64;
    const float* qh = qs + h * 64;
    float s = 0.0f;
    for (int d = 0; d < 64; ++d) s += qh[d] * b2f(kr[d]);
    lg[t] = s * 0.125f;                  // 1/sqrt(64)
  }
  __syncthreads();

  if (t < 8) {
    float m = -__builtin_inff();
    for (int j = 0; j < LKV; ++j) m = fmaxf(m, lg[t * LKV + j]);
    float sum = 0.0f;
    for (int j = 0; j < LKV; ++j) {
      float e = expf(lg[t * LKV + j] - m);
      lg[t * LKV + j] = e; sum += e;
    }
    float inv = 1.0f / sum;
    for (int j = 0; j < LKV; ++j) lg[t * LKV + j] *= inv;
  }
  __syncthreads();

#pragma unroll
  for (int i = 0; i < 2; ++i) {
    int c = t + i * 256; int h = c >> 6;
    float acc = 0.0f;
    for (int j = 0; j < LKV; ++j)
      acc += lg[h * LKV + j] * b2f(vh[((size_t)row * LKV + j) * DMODEL + c]);
    ob[(size_t)row * DMODEL + c] = f2b(acc);
  }
}

// out = normalize( normalize(fine) + normalize(coarse) )
__global__ __launch_bounds__(256) void final_norm(
    const float* __restrict__ fine, const float* __restrict__ coarse,
    float* __restrict__ out)
{
  __shared__ float r1[256], r2[256];
  const int row = blockIdx.x, t = threadIdx.x;
  const size_t base = (size_t)row * DMODEL;
  float f0 = fine[base + t],   f1 = fine[base + t + 256];
  float c0 = coarse[base + t], c1 = coarse[base + t + 256];
  r1[t] = f0 * f0 + f1 * f1; r2[t] = c0 * c0 + c1 * c1;
  __syncthreads();
  for (int s = 128; s > 0; s >>= 1) {
    if (t < s) { r1[t] += r1[t + s]; r2[t] += r2[t + s]; }
    __syncthreads();
  }
  const float rf = rsqrtf(r1[0]);
  const float rc = rsqrtf(r2[0]);
  float a0 = f0 * rf + c0 * rc, a1 = f1 * rf + c1 * rc;
  __syncthreads();
  r1[t] = a0 * a0 + a1 * a1;
  __syncthreads();
  for (int s = 128; s > 0; s >>= 1) {
    if (t < s) r1[t] += r1[t + s];
    __syncthreads();
  }
  const float ra = rsqrtf(r1[0]);
  out[base + t] = a0 * ra; out[base + t + 256] = a1 * ra;
}

// ---------------------------------------------------------------------------
// Host orchestration
// ---------------------------------------------------------------------------
static inline void cvt(const float* src, bf16_t* dst, long n, hipStream_t s) {
  f32_to_bf16<<<(unsigned)((n + 255) / 256), 256, 0, s>>>(src, dst, n);
}
static inline void gemm(const bf16_t* A, const bf16_t* W, const float* bias,
                        float* Cf, bf16_t* Cb, int M, int N, int K, int relu,
                        hipStream_t s) {
  dim3 g((N + 63) / 64, M / 128);
  wmma_gemm_bf16<<<g, 256, 0, s>>>(A, W, bias, Cf, Cb, M, N, K, relu);
}

extern "C" void kernel_launch(void* const* d_in, const int* in_sizes, int n_in,
                              void* d_out, int out_size, void* d_ws,
                              size_t ws_size, hipStream_t stream)
{
  (void)in_sizes; (void)n_in; (void)out_size; (void)ws_size;

  // ---- inputs (dict flattening order of setup_inputs) ----
  const float* X        = (const float*)d_in[0];   // (4096,512)
  const float* CF       = (const float*)d_in[1];   // (10000,512)
  const float* TYPE_EMB = (const float*)d_in[2];   // (2,512)
  const float* POS_EMB  = (const float*)d_in[3];   // (50,512)
  const float* CLS_EMB  = (const float*)d_in[4];   // (1,512)
  const float* REGION_W = (const float*)d_in[5];   // (512,512)
  const float* REGION_B = (const float*)d_in[6];   // (512,)
  // layers start at index 7, 26 arrays each:
  //  +0 saWq +1 sabq +2 saWk +3 sabk +4 saWv +5 sabv +6 saWo +7 sabo
  //  +8 caWq +9 cabq +10 caWk +11 cabk +12 caWv +13 cabv +14 caWo +15 cabo
  //  +16 W1 +17 bf1 +18 W2 +19 bf2 +20 g1 +21 b1 +22 g2 +23 b2 +24 g3 +25 b3

  // ---- workspace bump allocator (256B aligned) ----
  size_t off = 0;
  auto alloc = [&](size_t bytes) -> char* {
    off = (off + 255) & ~(size_t)255;
    char* p = (char*)d_ws + off;
    off += bytes;
    return p;
  };

  bf16_t* xb  = (bf16_t*)alloc((size_t)BB * DMODEL * 2);
  bf16_t* cfb = (bf16_t*)alloc((size_t)NCONC * DMODEL * 2);
  bf16_t* regWb = (bf16_t*)alloc((size_t)DMODEL * DMODEL * 2);

  bf16_t *saWv[NLAYER], *saWo[NLAYER], *caWq[NLAYER], *caWk[NLAYER],
         *caWv[NLAYER], *caWo[NLAYER], *W1b[NLAYER], *W2b[NLAYER];
  for (int l = 0; l < NLAYER; ++l) {
    saWv[l] = (bf16_t*)alloc((size_t)DMODEL * DMODEL * 2);
    saWo[l] = (bf16_t*)alloc((size_t)DMODEL * DMODEL * 2);
    caWq[l] = (bf16_t*)alloc((size_t)DMODEL * DMODEL * 2);
    caWk[l] = (bf16_t*)alloc((size_t)DMODEL * DMODEL * 2);
    caWv[l] = (bf16_t*)alloc((size_t)DMODEL * DMODEL * 2);
    caWo[l] = (bf16_t*)alloc((size_t)DMODEL * DMODEL * 2);
    W1b[l]  = (bf16_t*)alloc((size_t)DFF * DMODEL * 2);
    W2b[l]  = (bf16_t*)alloc((size_t)DMODEL * DFF * 2);
  }

  // union region: scores (164 MB) lifetime ends before kh/vh (214 MB each)
  const size_t per_kvh = (size_t)BB * LKV * DMODEL * 2;
  char* uni = alloc(2 * per_kvh);
  float*  scores = (float*)uni;
  bf16_t* khb = (bf16_t*)uni;
  bf16_t* vhb = (bf16_t*)(uni + per_kvh);

  bf16_t* kvb   = (bf16_t*)alloc(per_kvh);
  int*    idx   = (int*)  alloc((size_t)BB * TOPK * 4);
  float*  wsoft = (float*)alloc((size_t)BB * TOPK * 4);
  float*  tgtf  = (float*)alloc((size_t)BB * DMODEL * 4);
  bf16_t* tgtb  = (bf16_t*)alloc((size_t)BB * DMODEL * 2);
  float*  tmpAf = (float*)alloc((size_t)BB * DMODEL * 4);
  bf16_t* tmpAb = (bf16_t*)alloc((size_t)BB * DMODEL * 2);
  float*  tmpBf = (float*)alloc((size_t)BB * DMODEL * 4);
  float*  qf    = (float*)alloc((size_t)BB * DMODEL * 4);
  bf16_t* ob    = (bf16_t*)alloc((size_t)BB * DMODEL * 2);
  bf16_t* hb    = (bf16_t*)alloc((size_t)BB * DFF * 2);
  float*  coarsef = (float*)alloc((size_t)BB * DMODEL * 4);

  // ---- weight / input conversion to bf16 ----
  cvt(X, xb, (long)BB * DMODEL, stream);
  cvt(CF, cfb, (long)NCONC * DMODEL, stream);
  cvt(REGION_W, regWb, (long)DMODEL * DMODEL, stream);
  for (int l = 0; l < NLAYER; ++l) {
    const int base = 7 + 26 * l;
    cvt((const float*)d_in[base + 4],  saWv[l], (long)DMODEL * DMODEL, stream);
    cvt((const float*)d_in[base + 6],  saWo[l], (long)DMODEL * DMODEL, stream);
    cvt((const float*)d_in[base + 8],  caWq[l], (long)DMODEL * DMODEL, stream);
    cvt((const float*)d_in[base + 10], caWk[l], (long)DMODEL * DMODEL, stream);
    cvt((const float*)d_in[base + 12], caWv[l], (long)DMODEL * DMODEL, stream);
    cvt((const float*)d_in[base + 14], caWo[l], (long)DMODEL * DMODEL, stream);
    cvt((const float*)d_in[base + 16], W1b[l],  (long)DFF * DMODEL, stream);
    cvt((const float*)d_in[base + 18], W2b[l],  (long)DMODEL * DFF, stream);
  }

  // ---- concept scoring, top-k, kv construction ----
  gemm(xb, cfb, nullptr, scores, nullptr, BB, NCONC, DMODEL, 0, stream);
  topk_softmax<<<BB, 256, 0, stream>>>(scores, idx, wsoft);
  build_kv<<<BB * LKV, 256, 0, stream>>>(X, CF, TYPE_EMB, POS_EMB, idx, wsoft,
                                         kvb);
  broadcast_cls<<<BB, 256, 0, stream>>>(CLS_EMB, tgtf, tgtb);

  // ---- 6 decoder layers ----
  for (int l = 0; l < NLAYER; ++l) {
    const int base = 7 + 26 * l;
    const float* sabv = (const float*)d_in[base + 5];
    const float* sabo = (const float*)d_in[base + 7];
    const float* cabq = (const float*)d_in[base + 9];
    const float* cabk = (const float*)d_in[base + 11];
    const float* cabv = (const float*)d_in[base + 13];
    const float* cabo = (const float*)d_in[base + 15];
    const float* bf1  = (const float*)d_in[base + 17];
    const float* bf2  = (const float*)d_in[base + 19];
    const float* g1 = (const float*)d_in[base + 20];
    const float* b1 = (const float*)d_in[base + 21];
    const float* g2 = (const float*)d_in[base + 22];
    const float* b2 = (const float*)d_in[base + 23];
    const float* g3 = (const float*)d_in[base + 24];
    const float* b3 = (const float*)d_in[base + 25];

    // self-attention on length-1 target == Wo(Wv t + bv) + bo (exact)
    gemm(tgtb, saWv[l], sabv, tmpAf, tmpAb, BB, DMODEL, DMODEL, 0, stream);
    gemm(tmpAb, saWo[l], sabo, tmpBf, nullptr, BB, DMODEL, DMODEL, 0, stream);
    ln_residual<<<BB, 256, 0, stream>>>(tmpBf, tgtf, g1, b1, tgtf, tgtb);

    // cross-attention: q proj; K/V projections over all 51 kv tokens (the
    // dominant GEMMs, M = 4096*51); fused softmax-attention; out proj.
    gemm(tgtb, caWq[l], cabq, qf, nullptr, BB, DMODEL, DMODEL, 0, stream);
    gemm(kvb, caWk[l], cabk, nullptr, khb, BB * LKV, DMODEL, DMODEL, 0, stream);
    gemm(kvb, caWv[l], cabv, nullptr, vhb, BB * LKV, DMODEL, DMODEL, 0, stream);
    cross_attn<<<BB, 256, 0, stream>>>(qf, khb, vhb, ob);
    gemm(ob, caWo[l], cabo, tmpBf, nullptr, BB, DMODEL, DMODEL, 0, stream);
    ln_residual<<<BB, 256, 0, stream>>>(tmpBf, tgtf, g2, b2, tgtf, tgtb);

    // FFN: relu(t W1^T + bf1) W2^T + bf2
    gemm(tgtb, W1b[l], bf1, nullptr, hb, BB, DFF, DMODEL, 1, stream);
    gemm(hb, W2b[l], bf2, tmpBf, nullptr, BB, DMODEL, DFF, 0, stream);
    ln_residual<<<BB, 256, 0, stream>>>(tmpBf, tgtf, g3, b3, tgtf, tgtb);
  }

  // ---- region head + final double-normalize ----
  gemm(xb, regWb, REGION_B, coarsef, nullptr, BB, DMODEL, DMODEL, 0, stream);
  final_norm<<<BB, 256, 0, stream>>>(tgtf, coarsef, (float*)d_out);
}
